// Generator_Block_68487548502384
// MI455X (gfx1250) — matrix operand order repaired and checked by reading, not compiled, not run on hardware
//
#include <hip/hip_runtime.h>

typedef __bf16 bf16_t;
typedef bf16_t v16bf __attribute__((ext_vector_type(16)));
typedef float  v8f   __attribute__((ext_vector_type(8)));
typedef unsigned int v4u __attribute__((ext_vector_type(4)));

union Frag { v16bf v; v4u u[2]; };

__device__ __forceinline__ unsigned short f2bf(float f) {
  unsigned u = __float_as_uint(f);
  u += 0x7FFFu + ((u >> 16) & 1u);           // round to nearest even
  return (unsigned short)(u >> 16);
}

// ---------------------------------------------------------------------------
// Styles: s[b,co] = (w[b,:] @ fw) * scale + fb[co]      (16x512 @ 512x512)
// ---------------------------------------------------------------------------
__global__ __launch_bounds__(512) void styles_kernel(
    const float* __restrict__ w, const float* __restrict__ fw,
    const float* __restrict__ fb, float* __restrict__ sout, float scale)
{
  __shared__ float sw[512];
  const int b = blockIdx.x, co = threadIdx.x;
  sw[co] = w[b * 512 + co];
  __syncthreads();
  float acc = 0.f;
  for (int ci = 0; ci < 512; ++ci) acc += sw[ci] * fw[ci * 512 + co];
  sout[b * 512 + co] = acc * scale + fb[co];
}

// W2[ci,co] = sum_taps conv_w[tap,ci,co]^2
__global__ __launch_bounds__(256) void w2_kernel(
    const float* __restrict__ cw, float* __restrict__ w2)
{
  const int idx = blockIdx.x * 256 + threadIdx.x;     // 512*512
  const int co = idx & 511, ci = idx >> 9;
  float s = 0.f;
  for (int tap = 0; tap < 9; ++tap) {
    float v = cw[((size_t)tap * 512 + ci) * 512 + co];
    s += v * v;
  }
  w2[idx] = s;                                        // [ci][co]
}

// d[b,co] = rsqrt(wscale^2 * sum_ci s^2 * W2 + 1e-8)
__global__ __launch_bounds__(512) void demod_kernel(
    const float* __restrict__ s, const float* __restrict__ w2,
    float* __restrict__ d)
{
  __shared__ float ss[512];
  const int b = blockIdx.x, co = threadIdx.x;
  const float sv = s[b * 512 + co];
  ss[co] = sv * sv;
  __syncthreads();
  float acc = 0.f;
  for (int ci = 0; ci < 512; ++ci) acc += ss[ci] * w2[ci * 512 + co];
  d[b * 512 + co] = rsqrtf(acc * (1.0f / 4608.0f) + 1e-8f);
}

// wT[tap][co][ci] = bf16(conv_w[tap][ci][co] * wscale)
__global__ __launch_bounds__(256) void wcast_kernel(
    const float* __restrict__ cw, unsigned short* __restrict__ wT, float wscale)
{
  const int idx = blockIdx.x * 256 + threadIdx.x;     // 9*512*512
  const int ci = idx & 511;
  const int co = (idx >> 9) & 511;
  const int tap = idx >> 18;
  const float v = cw[((size_t)tap * 512 + ci) * 512 + co] * wscale;
  wT[idx] = f2bf(v);
}

// Style-modulate conv1 input: z1[b,y,x,ci] = bf16(x * s1), 32x32, NO stuffing
__global__ __launch_bounds__(256) void premod_kernel(
    const float* __restrict__ xin, const float* __restrict__ s1,
    unsigned short* __restrict__ z)
{
  const size_t idx = (size_t)blockIdx.x * 256 + threadIdx.x;  // 16*32*32*512
  const int ci = idx & 511;
  const int b  = (int)(idx >> 19);
  z[idx] = f2bf(xin[idx] * s1[b * 512 + ci]);
}

// ---------------------------------------------------------------------------
// Conv-transpose (up 2x) via parity decomposition, bf16 WMMA.
// Block: 256 thr = 8 waves. Tile: 16x16 OUTPUT pixels x 64 couts.
// Each parity class (cy,cx) has 64 pixels (8x8 grid of q) = 4 M-tiles.
// Wave w: m-tile j = w>>1 in EVERY class, cout half h = w&1 -> 8 accumulators.
// Taps per class: (e,e):{(1,1)}  (e,o):{(1,0),(1,2)}  (o,e):{(0,1),(2,1)}
//                 (o,o):{(0,0),(0,2),(2,0),(2,2)}; input offset d = 0/+1.
// out[b, 2q+c + T0, co] = d1 * sum taps ; input patch = 9x9 of 32x32 source.
// ---------------------------------------------------------------------------
__global__ __launch_bounds__(256) void convT_wmma_kernel(
    const unsigned short* __restrict__ zin,   // [16][32][32][512] bf16
    const unsigned short* __restrict__ wT,    // [9][512co][512ci] bf16
    const float* __restrict__ dmod,           // [16][512]
    float* __restrict__ out)                  // [16][64][64][512]
{
  __shared__ __align__(16) unsigned short sIn[9 * 9 * 32];     // 5184 B
  __shared__ __align__(16) unsigned short sW[9 * 64 * 32];     // 36 KB

  const int tile = blockIdx.x;                // b*16 + ty*4 + tx
  const int b  = tile >> 4;
  const int ty = (tile >> 2) & 3;
  const int tx = tile & 3;
  const int coBlk = blockIdx.y << 6;

  const int tid  = threadIdx.x;
  const int lane = tid & 31;
  const int wave = tid >> 5;
  const int j = wave >> 1;                    // M tile within each class
  const int h = wave & 1;                     // cout half

  const v8f vz = {0.f, 0.f, 0.f, 0.f, 0.f, 0.f, 0.f, 0.f};
  v8f acc[4][2];
  #pragma unroll
  for (int c = 0; c < 4; ++c) { acc[c][0] = vz; acc[c][1] = vz; }

  const int am  = lane & 15;
  const int akh = (lane >> 4) << 4;
  const int pix = j * 16 + am;
  const int qy = pix >> 3, qx = pix & 7;
  const int bn = lane & 15;
  const int bk = (lane >> 4) << 4;

  const int it1 = tid + 256;                  // second input item (if < 324)
  v4u rIn0, rIn1;
  v4u rW[9];

  // --- prefetch chunk kc into registers ---
  auto load_chunk = [&](int kc) {
    {
      const int point = tid >> 2, sub = tid & 3;
      const int py = point / 9, px = point % 9;
      const int iy = ty * 8 + py, ix = tx * 8 + px;
      v4u v = {0u, 0u, 0u, 0u};
      if (iy < 32 && ix < 32)
        v = *(const v4u*)(zin + (((size_t)b * 32 + iy) * 32 + ix) * 512 + kc * 32 + sub * 8);
      rIn0 = v;
    }
    if (it1 < 324) {
      const int point = it1 >> 2, sub = it1 & 3;
      const int py = point / 9, px = point % 9;
      const int iy = ty * 8 + py, ix = tx * 8 + px;
      v4u v = {0u, 0u, 0u, 0u};
      if (iy < 32 && ix < 32)
        v = *(const v4u*)(zin + (((size_t)b * 32 + iy) * 32 + ix) * 512 + kc * 32 + sub * 8);
      rIn1 = v;
    }
    #pragma unroll
    for (int jj = 0; jj < 9; ++jj) {
      const int it  = tid + (jj << 8);
      const int tap = it >> 8;
      const int rem = it & 255;
      const int co  = rem >> 2, sub = rem & 3;
      rW[jj] = *(const v4u*)(wT + ((size_t)tap * 512 + (coBlk + co)) * 512 +
                             kc * 32 + sub * 8);
    }
  };

  load_chunk(0);

  for (int kc = 0; kc < 16; ++kc) {
    ((v4u*)sIn)[tid] = rIn0;
    if (it1 < 324) ((v4u*)sIn)[it1] = rIn1;
    #pragma unroll
    for (int jj = 0; jj < 9; ++jj) ((v4u*)sW)[tid + (jj << 8)] = rW[jj];
    __syncthreads();
    if (kc + 1 < 16) load_chunk(kc + 1);      // overlap with WMMA below

    Frag A, B0, B1;
#define DO_TAP(CLS, TAP, DY, DX)                                              \
    { const unsigned short* ain = &sIn[((qy + (DY)) * 9 + (qx + (DX))) * 32 + akh]; \
      A.u[0] = *(const v4u*)(ain); A.u[1] = *(const v4u*)(ain + 8);           \
      const unsigned short* bp0 = &sW[((TAP) * 64 + h * 32 + bn) * 32 + bk];  \
      B0.u[0] = *(const v4u*)(bp0); B0.u[1] = *(const v4u*)(bp0 + 8);         \
      const unsigned short* bp1 = bp0 + 512;                                  \
      B1.u[0] = *(const v4u*)(bp1); B1.u[1] = *(const v4u*)(bp1 + 8);         \
      acc[CLS][0] = __builtin_amdgcn_wmma_f32_16x16x32_bf16(                  \
          false, A.v, false, B0.v, (short)0, acc[CLS][0], false, false);      \
      acc[CLS][1] = __builtin_amdgcn_wmma_f32_16x16x32_bf16(                  \
          false, A.v, false, B1.v, (short)0, acc[CLS][1], false, false); }

    DO_TAP(0, 4, 0, 0)
    DO_TAP(1, 3, 0, 0) DO_TAP(1, 5, 0, 1)
    DO_TAP(2, 1, 0, 0) DO_TAP(2, 7, 1, 0)
    DO_TAP(3, 0, 0, 0) DO_TAP(3, 2, 0, 1) DO_TAP(3, 6, 1, 0) DO_TAP(3, 8, 1, 1)
#undef DO_TAP
    __syncthreads();
  }

  #pragma unroll
  for (int cls = 0; cls < 4; ++cls) {
    const int cy = cls >> 1, cx = cls & 1;
    #pragma unroll
    for (int r = 0; r < 8; ++r) {
      const int mm = r + ((lane >> 4) << 3);
      const int nn = lane & 15;
      const int pix2 = j * 16 + mm;
      const int qy2 = pix2 >> 3, qx2 = pix2 & 7;
      const int oy = ty * 16 + 2 * qy2 + cy;
      const int ox = tx * 16 + 2 * qx2 + cx;
      const int co0 = coBlk + h * 32 + nn;
      const size_t base = (((size_t)b * 64 + oy) * 64 + ox) * 512;
      out[base + co0]      = acc[cls][0][r] * dmod[b * 512 + co0];
      out[base + co0 + 16] = acc[cls][1][r] * dmod[b * 512 + co0 + 16];
    }
  }
}

// ---------------------------------------------------------------------------
// Conv2: implicit-GEMM 3x3 stride-1, bf16 WMMA, fused demod+noise+bias+lrelu.
// Block: 8x8 output pixels (M=64) x 64 couts; wave: 16 rows x 32 couts.
// ---------------------------------------------------------------------------
__global__ __launch_bounds__(256) void conv3x3_wmma_kernel(
    const unsigned short* __restrict__ zin,   // [16][64][64][512] bf16 (pre-mod)
    const unsigned short* __restrict__ wT,    // [9][512co][512ci] bf16
    const float* __restrict__ dmod,           // [16][512]
    const float* __restrict__ noise,          // [16][64][64]
    const float* __restrict__ snoise,         // [512]
    const float* __restrict__ bias,           // [512]
    float* __restrict__ out)                  // [16][64][64][512]
{
  __shared__ __align__(16) unsigned short sIn[10 * 10 * 32];   // 6.25 KB
  __shared__ __align__(16) unsigned short sW[9 * 64 * 32];     // 36 KB

  const int tile = blockIdx.x;                // b*64 + ty*8 + tx
  const int b  = tile >> 6;
  const int ty = (tile >> 3) & 7;
  const int tx = tile & 7;
  const int coBlk = blockIdx.y << 6;

  const int tid  = threadIdx.x;
  const int lane = tid & 31;
  const int wave = tid >> 5;
  const int wm = wave & 3;
  const int wn = wave >> 2;

  v8f acc0 = {0.f, 0.f, 0.f, 0.f, 0.f, 0.f, 0.f, 0.f};
  v8f acc1 = acc0;

  const int am   = lane & 15;
  const int akh  = (lane >> 4) << 4;
  const int apix = wm * 16 + am;
  const int apy = apix >> 3, apx = apix & 7;
  const int bn = lane & 15;
  const int bk = (lane >> 4) << 4;

  const int it1 = tid + 256;                  // second input item (if < 400)
  v4u rIn0, rIn1;
  v4u rW[9];

  auto load_chunk = [&](int kc) {
    {
      const int point = tid >> 2, sub = tid & 3;
      const int py = point / 10, px = point % 10;
      const int iy = ty * 8 - 1 + py, ix = tx * 8 - 1 + px;
      v4u v = {0u, 0u, 0u, 0u};
      if ((unsigned)iy < 64u && (unsigned)ix < 64u)
        v = *(const v4u*)(zin + (((size_t)b * 64 + iy) * 64 + ix) * 512 + kc * 32 + sub * 8);
      rIn0 = v;
    }
    if (it1 < 400) {
      const int point = it1 >> 2, sub = it1 & 3;
      const int py = point / 10, px = point % 10;
      const int iy = ty * 8 - 1 + py, ix = tx * 8 - 1 + px;
      v4u v = {0u, 0u, 0u, 0u};
      if ((unsigned)iy < 64u && (unsigned)ix < 64u)
        v = *(const v4u*)(zin + (((size_t)b * 64 + iy) * 64 + ix) * 512 + kc * 32 + sub * 8);
      rIn1 = v;
    }
    #pragma unroll
    for (int jj = 0; jj < 9; ++jj) {
      const int it  = tid + (jj << 8);
      const int tap = it >> 8;
      const int rem = it & 255;
      const int co  = rem >> 2, sub = rem & 3;
      rW[jj] = *(const v4u*)(wT + ((size_t)tap * 512 + (coBlk + co)) * 512 +
                             kc * 32 + sub * 8);
    }
  };

  load_chunk(0);

  for (int kc = 0; kc < 16; ++kc) {
    ((v4u*)sIn)[tid] = rIn0;
    if (it1 < 400) ((v4u*)sIn)[it1] = rIn1;
    #pragma unroll
    for (int jj = 0; jj < 9; ++jj) ((v4u*)sW)[tid + (jj << 8)] = rW[jj];
    __syncthreads();
    if (kc + 1 < 16) load_chunk(kc + 1);      // overlap with WMMA below

    #pragma unroll
    for (int tap = 0; tap < 9; ++tap) {
      const int dy = tap / 3, dx = tap % 3;
      Frag A, B0, B1;
      const unsigned short* ain = &sIn[((apy + dy) * 10 + (apx + dx)) * 32 + akh];
      A.u[0] = *(const v4u*)(ain);
      A.u[1] = *(const v4u*)(ain + 8);
      const unsigned short* bp0 = &sW[(tap * 64 + wn * 32 + bn) * 32 + bk];
      B0.u[0] = *(const v4u*)(bp0);
      B0.u[1] = *(const v4u*)(bp0 + 8);
      const unsigned short* bp1 = bp0 + 512;
      B1.u[0] = *(const v4u*)(bp1);
      B1.u[1] = *(const v4u*)(bp1 + 8);
      acc0 = __builtin_amdgcn_wmma_f32_16x16x32_bf16(false, A.v, false, B0.v,
                                                     (short)0, acc0, false, false);
      acc1 = __builtin_amdgcn_wmma_f32_16x16x32_bf16(false, A.v, false, B1.v,
                                                     (short)0, acc1, false, false);
    }
    __syncthreads();
  }

  #pragma unroll
  for (int r = 0; r < 8; ++r) {
    const int mm = r + ((lane >> 4) << 3);
    const int nn = lane & 15;
    const int pix = wm * 16 + mm;
    const int oy = ty * 8 + (pix >> 3), ox = tx * 8 + (pix & 7);
    const int co0 = coBlk + wn * 32 + nn;
    const size_t base = (((size_t)b * 64 + oy) * 64 + ox) * 512;
    const float nv = noise[((size_t)b * 64 + oy) * 64 + ox];
    float v0 = acc0[r] * dmod[b * 512 + co0] + snoise[co0] * nv + bias[co0];
    float v1 = acc1[r] * dmod[b * 512 + co0 + 16] + snoise[co0 + 16] * nv + bias[co0 + 16];
    v0 = v0 > 0.f ? v0 : 0.2f * v0;
    v1 = v1 > 0.f ? v1 : 0.2f * v1;
    out[base + co0]      = v0;
    out[base + co0 + 16] = v1;
  }
}

// ---------------------------------------------------------------------------
// Blur(4x4, pad 1/2) + noise + bias + lrelu + s2-premod -> bf16 conv2 input
// ---------------------------------------------------------------------------
__global__ __launch_bounds__(256) void blurfuse_kernel(
    const float* __restrict__ y1, const float* __restrict__ noise1,
    const float* __restrict__ sn1, const float* __restrict__ bias1,
    const float* __restrict__ s2, unsigned short* __restrict__ z)
{
  const size_t idx = (size_t)blockIdx.x * 256 + threadIdx.x;  // 16*64*64*512
  const int c = idx & 511;
  const int x = (idx >> 9) & 63;
  const int y = (idx >> 15) & 63;
  const int b = (int)(idx >> 21);
  const float g[4] = {0.25f, 0.75f, 0.75f, 0.25f};            // f*2, gain folded
  float acc = 0.f;
  #pragma unroll
  for (int i = 0; i < 4; ++i) {
    const int yy = y - 1 + i;
    if ((unsigned)yy < 64u) {
      #pragma unroll
      for (int jj = 0; jj < 4; ++jj) {
        const int xx = x - 1 + jj;
        if ((unsigned)xx < 64u)
          acc += g[i] * g[jj] * y1[(((size_t)b * 64 + yy) * 64 + xx) * 512 + c];
      }
    }
  }
  float v = acc + sn1[c] * noise1[((size_t)b * 64 + y) * 64 + x] + bias1[c];
  v = v > 0.f ? v : 0.2f * v;
  z[idx] = f2bf(v * s2[b * 512 + c]);
}

// toRGB: 1x1, 512->3, no demod; one wave per pixel, shuffle reduction
__global__ __launch_bounds__(256) void rgb_kernel(
    const float* __restrict__ x2, const float* __restrict__ sr,
    const float* __restrict__ cw, const float* __restrict__ br,
    float* __restrict__ rgb)
{
  const int wave = threadIdx.x >> 5, lane = threadIdx.x & 31;
  const size_t p = (size_t)blockIdx.x * 8 + wave;             // 16*64*64 pixels
  const int b = (int)(p >> 12);
  const float wsr = 0.04419417382415922f;                     // 1/sqrt(512)
  float a0 = 0.f, a1 = 0.f, a2 = 0.f;
  for (int ci = lane; ci < 512; ci += 32) {
    const float xv = x2[p * 512 + ci] * (sr[b * 512 + ci] * wsr);
    a0 += xv * cw[ci * 3 + 0];
    a1 += xv * cw[ci * 3 + 1];
    a2 += xv * cw[ci * 3 + 2];
  }
  #pragma unroll
  for (int off = 16; off > 0; off >>= 1) {
    a0 += __shfl_xor(a0, off);
    a1 += __shfl_xor(a1, off);
    a2 += __shfl_xor(a2, off);
  }
  if (lane == 0) {
    float v0 = a0 + br[0]; v0 = v0 > 0.f ? v0 : 0.2f * v0;
    float v1 = a1 + br[1]; v1 = v1 > 0.f ? v1 : 0.2f * v1;
    float v2 = a2 + br[2]; v2 = v2 > 0.f ? v2 : 0.2f * v2;
    rgb[p * 3 + 0] = v0;
    rgb[p * 3 + 1] = v1;
    rgb[p * 3 + 2] = v2;
  }
}

// ---------------------------------------------------------------------------
extern "C" void kernel_launch(void* const* d_in, const int* in_sizes, int n_in,
                              void* d_out, int out_size, void* d_ws, size_t ws_size,
                              hipStream_t stream) {
  (void)in_sizes; (void)n_in; (void)out_size; (void)ws_size;
  const float* x       = (const float*)d_in[0];
  const float* w       = (const float*)d_in[1];
  const float* noise1  = (const float*)d_in[2];
  const float* noise2  = (const float*)d_in[3];
  const float* fcl1_w  = (const float*)d_in[4];
  const float* fcl1_b  = (const float*)d_in[5];
  const float* conv1_w = (const float*)d_in[6];
  const float* sn1     = (const float*)d_in[7];
  const float* bias1   = (const float*)d_in[8];
  const float* fcl2_w  = (const float*)d_in[9];
  const float* fcl2_b  = (const float*)d_in[10];
  const float* conv2_w = (const float*)d_in[11];
  const float* sn2     = (const float*)d_in[12];
  const float* bias2   = (const float*)d_in[13];
  const float* fclr_w  = (const float*)d_in[14];
  const float* fclr_b  = (const float*)d_in[15];
  const float* convr_w = (const float*)d_in[16];
  const float* biasr   = (const float*)d_in[17];

  char* ws = (char*)d_ws;
  float* s1  = (float*)(ws + 0);
  float* s2  = (float*)(ws + 32768);
  float* sr  = (float*)(ws + 65536);
  float* d1  = (float*)(ws + 98304);
  float* d2  = (float*)(ws + 131072);
  float* W2a = (float*)(ws + 163840);
  float* W2b = (float*)(ws + 163840 + 1048576);
  unsigned short* wT1 = (unsigned short*)(ws + 163840 + 2 * 1048576);
  unsigned short* wT2 = wT1 + (size_t)9 * 512 * 512;
  unsigned short* z   = wT2 + (size_t)9 * 512 * 512;   // 64 MB region
  float* y1 = (float*)((char*)z + (size_t)16 * 64 * 64 * 512 * 2);

  float* x2  = (float*)d_out;                       // [16][64][64][512]
  float* rgb = x2 + (size_t)16 * 64 * 64 * 512;     // [16][64][64][3]

  const float fscale = 0.04419417382415922f;        // 1/sqrt(512)
  const float wscale = 0.014731391274719739f;       // 1/sqrt(9*512)

  styles_kernel<<<16, 512, 0, stream>>>(w, fcl1_w, fcl1_b, s1, fscale);
  styles_kernel<<<16, 512, 0, stream>>>(w, fcl2_w, fcl2_b, s2, fscale);
  styles_kernel<<<16, 512, 0, stream>>>(w, fclr_w, fclr_b, sr, fscale);
  w2_kernel<<<1024, 256, 0, stream>>>(conv1_w, W2a);
  w2_kernel<<<1024, 256, 0, stream>>>(conv2_w, W2b);
  demod_kernel<<<16, 512, 0, stream>>>(s1, W2a, d1);
  demod_kernel<<<16, 512, 0, stream>>>(s2, W2b, d2);
  wcast_kernel<<<9216, 256, 0, stream>>>(conv1_w, wT1, wscale);
  wcast_kernel<<<9216, 256, 0, stream>>>(conv2_w, wT2, wscale);
  premod_kernel<<<32768, 256, 0, stream>>>(x, s1, z);     // z1: 32x32 bf16

  dim3 tgrid(256, 8);                                     // conv-transpose
  convT_wmma_kernel<<<tgrid, 256, 0, stream>>>(z, wT1, d1, y1);
  blurfuse_kernel<<<131072, 256, 0, stream>>>(y1, noise1, sn1, bias1, s2, z);
  dim3 cgrid(1024, 8);                                    // conv2
  conv3x3_wmma_kernel<<<cgrid, 256, 0, stream>>>(z, wT2, d2, noise2, sn2,
                                                 bias2, x2);
  rgb_kernel<<<8192, 256, 0, stream>>>(x2, sr, convr_w, biasr, rgb);
}
